// AttentionRNN_74457553044042
// MI455X (gfx1250) — matrix operand order repaired and checked by reading, not compiled, not run on hardware
//
#include <hip/hip_runtime.h>
#include <math.h>
#include <stdint.h>

typedef __attribute__((ext_vector_type(16))) __bf16 v16bf;
typedef __attribute__((ext_vector_type(8)))  float  v8f;

#define B_  64
#define T_  2048
#define H_  256
#define IN_ 512
#define PITCH_U 132   // 64-row LDS tile pitch in dwords: 128 data + 4 pad (bank-conflict free)

__device__ __forceinline__ uint32_t pack_bf16(float a, float b) {
    // round-to-nearest-even fp32 -> bf16, pack two into one dword (lo = a, hi = b)
    uint32_t ua = __float_as_uint(a);
    uint32_t ub = __float_as_uint(b);
    ua += 0x7FFFu + ((ua >> 16) & 1u);
    ub += 0x7FFFu + ((ub >> 16) & 1u);
    return (ua >> 16) | (ub & 0xFFFF0000u);
}

// ---------------------------------------------------------------------------
// K0: Wa (256x256 f32, row-major) -> bf16 packed pairs in workspace
// ---------------------------------------------------------------------------
__global__ __launch_bounds__(256) void conv_wa_bf16_kernel(
    const float* __restrict__ src, uint32_t* __restrict__ dst) {
    int idx = blockIdx.x * 256 + threadIdx.x;          // 0..32767 pair index
    float2 f = *(const float2*)&src[(size_t)idx * 2];
    dst[idx] = pack_bf16(f.x, f.y);
}

// ---------------------------------------------------------------------------
// K1: GRU cell (PyTorch semantics) + pq = rnn_out @ Wq^T + bq
// One block per batch row, 256 threads (one per hidden unit).
// ---------------------------------------------------------------------------
__global__ __launch_bounds__(256) void gru_pq_kernel(
    const float* __restrict__ memory, const float* __restrict__ context,
    const float* __restrict__ h_in,   const float* __restrict__ W_ih,
    const float* __restrict__ W_hh,   const float* __restrict__ b_ih,
    const float* __restrict__ b_hh,   const float* __restrict__ Wq,
    const float* __restrict__ bq,     float* __restrict__ rnn_out,
    float* __restrict__ pq_ws) {
    __shared__ float sx[IN_];
    __shared__ float sh[H_];
    __shared__ float sr[H_];
    const int b = blockIdx.x, j = threadIdx.x;
    sx[j]       = memory[b * H_ + j];
    sx[H_ + j]  = context[b * H_ + j];
    sh[j]       = h_in[b * H_ + j];
    __syncthreads();

    const float* wr = &W_ih[(size_t)(0 * H_ + j) * IN_];
    const float* wz = &W_ih[(size_t)(1 * H_ + j) * IN_];
    const float* wn = &W_ih[(size_t)(2 * H_ + j) * IN_];
    float ir = b_ih[j], iz = b_ih[H_ + j], in_ = b_ih[2 * H_ + j];
    #pragma unroll 4
    for (int k = 0; k < IN_; ++k) {
        float x = sx[k];
        ir += x * wr[k]; iz += x * wz[k]; in_ += x * wn[k];
    }
    const float* ur = &W_hh[(size_t)(0 * H_ + j) * H_];
    const float* uz = &W_hh[(size_t)(1 * H_ + j) * H_];
    const float* un = &W_hh[(size_t)(2 * H_ + j) * H_];
    float hr = b_hh[j], hz = b_hh[H_ + j], hn = b_hh[2 * H_ + j];
    #pragma unroll 4
    for (int k = 0; k < H_; ++k) {
        float h = sh[k];
        hr += h * ur[k]; hz += h * uz[k]; hn += h * un[k];
    }
    float r = 1.0f / (1.0f + expf(-(ir + hr)));
    float z = 1.0f / (1.0f + expf(-(iz + hz)));
    float n = tanhf(in_ + r * hn);
    float o = (1.0f - z) * n + z * sh[j];
    rnn_out[b * H_ + j] = o;
    sr[j] = o;
    __syncthreads();

    const float* wq = &Wq[(size_t)j * H_];
    float acc = bq[j];
    #pragma unroll 4
    for (int k = 0; k < H_; ++k) acc += sr[k] * wq[k];
    pq_ws[b * H_ + j] = acc;
}

// ---------------------------------------------------------------------------
// K2: scores[b,t] = tanh(pq[b]+ba + ann[b,t,:]@Wa^T) . v  via bf16 WMMA
// Block = (b, 64-timestep tile), 256 threads = 8 waves.
// Wave w: M-tile = w>>1 (16 rows), N-half = w&1 (128 cols), 8 f32 accumulators.
// ---------------------------------------------------------------------------
__global__ __launch_bounds__(256) void scores_wmma_kernel(
    const float*    __restrict__ ann,   const uint32_t* __restrict__ wa_u,
    const float*    __restrict__ pq_ws, const float*    __restrict__ ba,
    const float*    __restrict__ v,     float*          __restrict__ scores) {
    __shared__ uint32_t sA[64 * PITCH_U];   // 64 x 256 bf16 annotation tile
    __shared__ float    s_pq[H_];           // pq + ba
    __shared__ float    s_v[H_];
    __shared__ float    sS[64];             // per-row score accumulators

    const int tid = threadIdx.x;
    const int b = blockIdx.y;
    const int t0 = blockIdx.x * 64;

    s_pq[tid] = pq_ws[b * H_ + tid] + ba[tid];
    s_v[tid]  = v[tid];
    if (tid < 64) sS[tid] = 0.0f;

    // Stage 64x256 fp32 -> bf16 into LDS (8192 packed pairs).
    const float* abase = &ann[((size_t)b * T_ + t0) * H_];
    #pragma unroll
    for (int i = 0; i < 32; ++i) {
        int idx = tid + i * 256;           // pair index
        int row = idx >> 7;                // /128 pairs per row
        int cp  = idx & 127;
        float2 f = *(const float2*)&abase[(size_t)row * H_ + cp * 2];
        sA[row * PITCH_U + cp] = pack_bf16(f.x, f.y);
    }
    __syncthreads();

    const int lane = tid & 31;
    const int wave = tid >> 5;
    const int mt   = wave >> 1;            // 0..3  (16-row tile)
    const int nh   = wave & 1;             // 0..1  (128-col half)
    const int r    = lane & 15;
    const int half = lane >> 4;

    v8f zero = {0.f, 0.f, 0.f, 0.f, 0.f, 0.f, 0.f, 0.f};
    v8f acc[8];
    #pragma unroll
    for (int i = 0; i < 8; ++i) acc[i] = zero;

    const uint32_t* arow = &sA[(mt * 16 + r) * PITCH_U];
    #pragma unroll
    for (int ks = 0; ks < 8; ++ks) {       // K = 256, 32 per step
        // A fragment, 16-bit A 16x32 layout: lanes 0-15 hold K 0-7 & 16-23,
        // lanes 16-31 hold K 8-15 & 24-31 (each as 4 packed dwords).
        union { v16bf v; uint32_t u[8]; } A;
        const uint32_t* ap = arow + ks * 16 + half * 4;
        #pragma unroll
        for (int q = 0; q < 4; ++q) { A.u[q] = ap[q]; A.u[4 + q] = ap[8 + q]; }

        #pragma unroll
        for (int nt = 0; nt < 8; ++nt) {
            // B fragment: column n = Wa row n (pa[t][n] = sum_k ann[k]*Wa[n][k]).
            // lanes 0-15: K 0-7 & 16-23, lanes 16-31: K 8-15 & 24-31 (same as A).
            int n = nh * 128 + nt * 16 + r;
            const uint32_t* bp = &wa_u[(size_t)n * 128 + ks * 16 + half * 4];
            union { v16bf v; uint32_t u[8]; } Bf;
            #pragma unroll
            for (int q = 0; q < 4; ++q) { Bf.u[q] = bp[q]; Bf.u[4 + q] = bp[8 + q]; }

            acc[nt] = __builtin_amdgcn_wmma_f32_16x16x32_bf16(
                false, A.v, false, Bf.v, (short)0, acc[nt], false, false);
        }
    }

    // Fused tanh(pa + pq + ba) * v, reduced over N.
    // C/D layout: VGPR g, lanes 0-15 -> row m0+g, lanes 16-31 -> row m0+8+g; N = n0 + (lane&15).
    float sg[8];
    #pragma unroll
    for (int g = 0; g < 8; ++g) sg[g] = 0.0f;
    #pragma unroll
    for (int nt = 0; nt < 8; ++nt) {
        union { v8f v; float f[8]; } C; C.v = acc[nt];
        int n = nh * 128 + nt * 16 + r;
        float pqa = s_pq[n], vv = s_v[n];
        #pragma unroll
        for (int g = 0; g < 8; ++g) sg[g] += tanhf(C.f[g] + pqa) * vv;
    }
    #pragma unroll
    for (int g = 0; g < 8; ++g) {
        float x = sg[g];
        x += __shfl_xor(x, 1, 32);
        x += __shfl_xor(x, 2, 32);
        x += __shfl_xor(x, 4, 32);
        x += __shfl_xor(x, 8, 32);         // reduce within each 16-lane half
        if (r == 0) atomicAdd(&sS[mt * 16 + half * 8 + g], x);
    }
    __syncthreads();
    if (tid < 64) scores[(size_t)b * T_ + t0 + tid] = sS[tid];
}

// ---------------------------------------------------------------------------
// K3: per-b softmax over T=2048 -> alignment; also zero the context row.
// ---------------------------------------------------------------------------
__global__ __launch_bounds__(256) void softmax_zero_kernel(
    const float* __restrict__ scores, float* __restrict__ out) {
    __shared__ float red[256];
    const int b = blockIdx.x, tid = threadIdx.x;
    float vals[8];
    float m = -3.0e38f;
    #pragma unroll
    for (int i = 0; i < 8; ++i) {
        vals[i] = scores[(size_t)b * T_ + tid + i * 256];
        m = fmaxf(m, vals[i]);
    }
    red[tid] = m; __syncthreads();
    for (int s = 128; s > 0; s >>= 1) {
        if (tid < s) red[tid] = fmaxf(red[tid], red[tid + s]);
        __syncthreads();
    }
    float gm = red[0];
    __syncthreads();
    float sum = 0.0f;
    #pragma unroll
    for (int i = 0; i < 8; ++i) { vals[i] = expf(vals[i] - gm); sum += vals[i]; }
    red[tid] = sum; __syncthreads();
    for (int s = 128; s > 0; s >>= 1) {
        if (tid < s) red[tid] += red[tid + s];
        __syncthreads();
    }
    float inv = 1.0f / red[0];
    float* align_out = out + 2 * B_ * H_;
    #pragma unroll
    for (int i = 0; i < 8; ++i)
        align_out[(size_t)b * T_ + tid + i * 256] = vals[i] * inv;
    out[B_ * H_ + b * H_ + tid] = 0.0f;    // zero context row for K4 atomics
}

// ---------------------------------------------------------------------------
// K4: new_context[b,h] += sum_t alignment[b,t] * annotations[b,t,h]
// Grid (T/128, B); coalesced over h; partial sums via global f32 atomics.
// ---------------------------------------------------------------------------
__global__ __launch_bounds__(256) void context_kernel(
    const float* __restrict__ ann, const float* __restrict__ align_in,
    float* __restrict__ ctx) {
    __shared__ float sa[128];
    const int b = blockIdx.y, c = blockIdx.x, tid = threadIdx.x;
    if (tid < 128) sa[tid] = align_in[(size_t)b * T_ + c * 128 + tid];
    __syncthreads();
    const float* ap = &ann[((size_t)b * T_ + c * 128) * H_ + tid];
    float acc = 0.0f;
    #pragma unroll 4
    for (int i = 0; i < 128; ++i) acc += sa[i] * ap[(size_t)i * H_];
    atomicAdd(&ctx[b * H_ + tid], acc);
}

// ---------------------------------------------------------------------------
extern "C" void kernel_launch(void* const* d_in, const int* in_sizes, int n_in,
                              void* d_out, int out_size, void* d_ws, size_t ws_size,
                              hipStream_t stream) {
    const float* memory      = (const float*)d_in[0];
    const float* context     = (const float*)d_in[1];
    const float* rnn_state   = (const float*)d_in[2];
    const float* annotations = (const float*)d_in[3];
    const float* W_ih        = (const float*)d_in[4];
    const float* W_hh        = (const float*)d_in[5];
    const float* b_ih        = (const float*)d_in[6];
    const float* b_hh        = (const float*)d_in[7];
    const float* Wq          = (const float*)d_in[8];
    const float* bq          = (const float*)d_in[9];
    const float* Wa          = (const float*)d_in[10];
    const float* ba          = (const float*)d_in[11];
    const float* v           = (const float*)d_in[12];

    char* ws = (char*)d_ws;
    uint32_t* wa_bf16   = (uint32_t*)ws;                    // 128 KB
    float*    pq_ws     = (float*)(ws + (128u << 10));      //  64 KB
    float*    scores_ws = (float*)(ws + (192u << 10));      // 512 KB

    float* out       = (float*)d_out;
    float* rnn_out   = out;                 // (B,H)
    float* ctx_out   = out + B_ * H_;       // (B,H)
    float* align_out = out + 2 * B_ * H_;   // (B,T)

    conv_wa_bf16_kernel<<<128, 256, 0, stream>>>(Wa, wa_bf16);
    gru_pq_kernel<<<B_, 256, 0, stream>>>(memory, context, rnn_state,
                                          W_ih, W_hh, b_ih, b_hh, Wq, bq,
                                          rnn_out, pq_ws);
    scores_wmma_kernel<<<dim3(T_ / 64, B_), 256, 0, stream>>>(
        annotations, wa_bf16, pq_ws, ba, v, scores_ws);
    softmax_zero_kernel<<<B_, 256, 0, stream>>>(scores_ws, out);
    context_kernel<<<dim3(T_ / 128, B_), 256, 0, stream>>>(
        annotations, align_out, ctx_out);
    (void)in_sizes; (void)n_in; (void)out_size; (void)ws_size;
}
